// RelMultiHeadAttention_25563645346438
// MI455X (gfx1250) — matrix operand order repaired and checked by reading, not compiled, not run on hardware
//
#include <hip/hip_runtime.h>
#include <hip/hip_bf16.h>

// ---- constants for this problem ----
#define BB 4
#define TT 1024
#define HH 1024
#define NHD 16
#define HDD 64
#define SCALE 0.125f  // HD^-0.5

// ---- vector types ----
typedef __attribute__((ext_vector_type(16))) __bf16 v16bf;
typedef __attribute__((ext_vector_type(8)))  __bf16 v8bf;
typedef __attribute__((ext_vector_type(4)))  __bf16 v4bf;
typedef __attribute__((ext_vector_type(8)))  float  v8f;
typedef __attribute__((ext_vector_type(4)))  unsigned int tdm_v4u;
typedef __attribute__((ext_vector_type(8)))  int          tdm_v8i;
typedef __attribute__((ext_vector_type(4)))  int          tdm_v4i;

// ---- WMMA fragment loaders (contiguous k order k = 16h + s in A and B, wave32) ----
// A 16x32 bf16: lanes 0-15 hold row M=lane, K 0..15;
//               lanes 16-31 hold row M=lane-16, K 16..31.
static __device__ __forceinline__ v16bf load_afrag(const __bf16* base, size_t row_stride, int lane) {
  const int row = lane & 15;
  const int kh  = (lane >> 4) << 4;  // 0 or 16
  const __bf16* rp = base + (size_t)row * row_stride;
  v8bf lo = *(const v8bf*)(rp + kh);
  v8bf hi = *(const v8bf*)(rp + kh + 8);
  v16bf a;
#pragma unroll
  for (int e = 0; e < 8; ++e) { a[e] = lo[e]; a[e + 8] = hi[e]; }
  return a;
}

// B 32x16 bf16: lane = N (mod 16); lanes 0-15 hold K 0..15, lanes 16-31 K 16..31.
// base points at (row n0 of the row-major "B-transposed" matrix, k0).
static __device__ __forceinline__ v16bf load_bfrag(const __bf16* base, size_t row_stride, int lane) {
  const int n  = lane & 15;
  const int kh = (lane >> 4) << 4;   // 0 or 16
  return *(const v16bf*)(base + (size_t)n * row_stride + kh);
}

static __device__ __forceinline__ v8f wmma_bf16(v16bf a, v16bf b, v8f c) {
  return __builtin_amdgcn_wmma_f32_16x16x32_bf16(false, a, false, b, (short)0, c, false, false);
}

// ---- fp32 -> bf16 convert (grid-stride, vectorized x4) ----
__global__ __launch_bounds__(256) void cvt_bf16_kernel(const float* __restrict__ in,
                                                       __bf16* __restrict__ out, int n4) {
  int i = blockIdx.x * blockDim.x + threadIdx.x;
  int stride = gridDim.x * blockDim.x;
  for (; i < n4; i += stride) {
    float4 v = ((const float4*)in)[i];
    v4bf o;
    o[0] = (__bf16)v.x; o[1] = (__bf16)v.y; o[2] = (__bf16)v.z; o[3] = (__bf16)v.w;
    ((v4bf*)out)[i] = o;
  }
}

// ---- QKV projection: q/k/v = x @ W^T + b, scattered into head-major layouts ----
// grid (16 Nblk, 32 Mblk, 3 mats), block 256 (8 waves); wave -> 16(M) x 64(N) tile.
__global__ __launch_bounds__(256) void qkv_kernel(
    const __bf16* __restrict__ xb,
    const __bf16* __restrict__ Wqb, const __bf16* __restrict__ Wkb, const __bf16* __restrict__ Wvb,
    const float* __restrict__ bq, const float* __restrict__ bk, const float* __restrict__ bv,
    __bf16* __restrict__ Q, __bf16* __restrict__ K, __bf16* __restrict__ Vt) {
  const int lane = threadIdx.x & 31, wave = threadIdx.x >> 5;
  const int z = blockIdx.z;
  const __bf16* W = (z == 0) ? Wqb : (z == 1) ? Wkb : Wvb;
  const float* bias = (z == 0) ? bq : (z == 1) ? bk : bv;
  const int Mtile = blockIdx.y * 128 + wave * 16;
  const int N0 = blockIdx.x * 64;
  v8f c[4] = {{0,0,0,0,0,0,0,0},{0,0,0,0,0,0,0,0},{0,0,0,0,0,0,0,0},{0,0,0,0,0,0,0,0}};
  for (int k0 = 0; k0 < HH; k0 += 32) {
    v16bf a = load_afrag(xb + (size_t)Mtile * HH + k0, HH, lane);
#pragma unroll
    for (int nt = 0; nt < 4; ++nt) {
      v16bf b = load_bfrag(W + (size_t)(N0 + nt * 16) * HH + k0, HH, lane);
      c[nt] = wmma_bf16(a, b, c[nt]);
    }
  }
  const int rl = lane & 15, hb = lane >> 4;
#pragma unroll
  for (int nt = 0; nt < 4; ++nt) {
    const int n = N0 + nt * 16 + rl;
    const float bvn = bias[n];
    const int h = n >> 6, d = n & 63;
#pragma unroll
    for (int r = 0; r < 8; ++r) {
      const int m = Mtile + r + 8 * hb;
      const int b_ = m >> 10, t = m & 1023;
      const float val = c[nt][r] + bvn;
      if (z == 2) {
        Vt[((size_t)(b_ * NHD + h) * HDD + d) * TT + t] = (__bf16)val;   // [bh][d][t]
      } else {
        const size_t idx = ((size_t)(b_ * NHD + h) * TT + t) * HDD + d;  // [bh][t][d]
        if (z == 0) Q[idx] = (__bf16)val; else K[idx] = (__bf16)val;
      }
    }
  }
}

// ---- scores = scale * Q K^T (+ mask) -> S fp32 [bh][i][j] ----
// grid (16 jblk, 8 iblk, 64 bh), block 256; wave -> 16(i) x 64(j).
__global__ __launch_bounds__(256) void scores_kernel(
    const __bf16* __restrict__ Q, const __bf16* __restrict__ K,
    const int* __restrict__ am, float* __restrict__ S) {
  const int lane = threadIdx.x & 31, wave = threadIdx.x >> 5;
  const int bh = blockIdx.z, b_ = bh >> 4;
  const __bf16* Qh = Q + (size_t)bh * TT * HDD;
  const __bf16* Kh = K + (size_t)bh * TT * HDD;
  const int itile = blockIdx.y * 128 + wave * 16;
  const int j0 = blockIdx.x * 64;
  v8f c[4] = {{0,0,0,0,0,0,0,0},{0,0,0,0,0,0,0,0},{0,0,0,0,0,0,0,0},{0,0,0,0,0,0,0,0}};
#pragma unroll
  for (int d0 = 0; d0 < HDD; d0 += 32) {
    v16bf a = load_afrag(Qh + (size_t)itile * HDD + d0, HDD, lane);
#pragma unroll
    for (int nt = 0; nt < 4; ++nt) {
      v16bf b = load_bfrag(Kh + (size_t)(j0 + nt * 16) * HDD + d0, HDD, lane);
      c[nt] = wmma_bf16(a, b, c[nt]);
    }
  }
  const int rl = lane & 15, hb = lane >> 4;
#pragma unroll
  for (int nt = 0; nt < 4; ++nt) {
    const int j = j0 + nt * 16 + rl;
    const bool keep = am[b_ * TT + j] != 0;
#pragma unroll
    for (int r = 0; r < 8; ++r) {
      const int i = itile + r + 8 * hb;
      float v = c[nt][r] * SCALE;
      if (!keep) v = -__builtin_inff();
      S[((size_t)bh * TT + i) * TT + j] = v;
    }
  }
}

// ---- position-bias: S[bh,i,j] += scale * sum_d Q[bh,i,d]*pb[i,j,d] ----
// For fixed i this is (64 bh x 64 d) x (64 d x T j) GEMM; pb read ONCE total.
// grid (8 jblk, 1024 i), block 256; pb tile (128 j x 64 d fp32 = 32KB, contiguous)
// staged in LDS by the Tensor Data Mover.
__global__ __launch_bounds__(256) void pbias_kernel(
    const __bf16* __restrict__ Q, const float* __restrict__ pb, float* __restrict__ S) {
  __shared__ float pbs[128 * 64];
  const int lane = threadIdx.x & 31, wave = threadIdx.x >> 5;
  const int i = blockIdx.y;
  const int jblk = blockIdx.x * 128;

  if (threadIdx.x < 32) {  // wave 0 issues the TDM descriptor
    const unsigned long long ga =
        (unsigned long long)(uintptr_t)(pb + ((size_t)i * TT + jblk) * HDD);
    const unsigned int lds = (unsigned int)(uintptr_t)(void*)pbs;
    tdm_v4u g0;
    g0[0] = 1u;                                   // count=1, user descriptor
    g0[1] = lds;                                  // lds_addr
    g0[2] = (unsigned int)ga;                     // global_addr[31:0]
    g0[3] = (unsigned int)((ga >> 32) & 0x1FFFFFFull) | (2u << 30);  // addr hi | type=2
    tdm_v8i g1;
    g1[0] = 0x20000;          // data_size=4B (code 2), no flags
    g1[1] = (int)0x20000000;  // tensor_dim0 lo16 (8192) << 16
    g1[2] = 0x10000;          // tensor_dim0 hi16=0 | tensor_dim1=1
    g1[3] = (int)0x20000000;  // tensor_dim1 hi=0 | tile_dim0=8192
    g1[4] = 1;                // tile_dim1=1, tile_dim2=0
    g1[5] = 8192;             // tensor_dim0_stride lo
    g1[6] = 0; g1[7] = 0;
    tdm_v4i gz4; gz4[0] = 0; gz4[1] = 0; gz4[2] = 0; gz4[3] = 0;
    tdm_v8i gz8;
#pragma unroll
    for (int e = 0; e < 8; ++e) gz8[e] = 0;
    __builtin_amdgcn_tensor_load_to_lds(g0, g1, gz4, gz4, gz8, 0);
    __builtin_amdgcn_s_wait_tensorcnt(0);
  }
  __syncthreads();

  const int jloc0 = wave * 16;         // this wave's 16-wide j strip
  const int rl = lane & 15;
  const int kh16 = (lane >> 4) << 4;
  v8f c[4] = {{0,0,0,0,0,0,0,0},{0,0,0,0,0,0,0,0},{0,0,0,0,0,0,0,0},{0,0,0,0,0,0,0,0}};
#pragma unroll
  for (int d0 = 0; d0 < HDD; d0 += 32) {
    const float* bp = &pbs[(jloc0 + rl) * HDD + d0 + kh16];
    v8f x0 = *(const v8f*)(bp);
    v8f x1 = *(const v8f*)(bp + 8);
    v16bf b;
#pragma unroll
    for (int e = 0; e < 8; ++e) { b[e] = (__bf16)x0[e]; b[e + 8] = (__bf16)x1[e]; }
#pragma unroll
    for (int mt = 0; mt < 4; ++mt) {  // M = bh rows 0..63 in four 16-row tiles
      v16bf a = load_afrag(Q + (size_t)(mt * 16) * (TT * HDD) + (size_t)i * HDD + d0,
                           (size_t)TT * HDD, lane);
      c[mt] = wmma_bf16(a, b, c[mt]);
    }
  }
  const int hb = lane >> 4;
  const int j = jblk + jloc0 + rl;
#pragma unroll
  for (int mt = 0; mt < 4; ++mt) {
#pragma unroll
    for (int r = 0; r < 8; ++r) {
      const int bh = mt * 16 + r + 8 * hb;
      const size_t idx = ((size_t)bh * TT + i) * TT + j;
      S[idx] += SCALE * c[mt][r];
    }
  }
}

// ---- row softmax over S, write bf16 probs P [bh][i][j] ----
__global__ __launch_bounds__(128) void softmax_kernel(const float* __restrict__ S,
                                                      __bf16* __restrict__ P) {
  const int row = blockIdx.x;  // 65536 rows (bh * 1024 + i)
  const float* src = S + (size_t)row * TT;
  __bf16* dst = P + (size_t)row * TT;
  const int t = threadIdx.x;   // 0..127, 8 elems each
  float4 a = ((const float4*)src)[t * 2];
  float4 b = ((const float4*)src)[t * 2 + 1];
  float e[8] = {a.x, a.y, a.z, a.w, b.x, b.y, b.z, b.w};
  float m = e[0];
#pragma unroll
  for (int k = 1; k < 8; ++k) m = fmaxf(m, e[k]);
#pragma unroll
  for (int off = 16; off > 0; off >>= 1) m = fmaxf(m, __shfl_xor(m, off, 32));
  __shared__ float redm[4], reds[4];
  const int wv = t >> 5, ln = t & 31;
  if (ln == 0) redm[wv] = m;
  __syncthreads();
  m = fmaxf(fmaxf(redm[0], redm[1]), fmaxf(redm[2], redm[3]));
  float s = 0.f;
#pragma unroll
  for (int k = 0; k < 8; ++k) { e[k] = __expf(e[k] - m); s += e[k]; }
#pragma unroll
  for (int off = 16; off > 0; off >>= 1) s += __shfl_xor(s, off, 32);
  if (ln == 0) reds[wv] = s;
  __syncthreads();
  s = reds[0] + reds[1] + reds[2] + reds[3];
  const float inv = 1.0f / s;
  v8bf o;
#pragma unroll
  for (int k = 0; k < 8; ++k) o[k] = (__bf16)(e[k] * inv);
  *(v8bf*)(dst + t * 8) = o;
}

// ---- O = P @ V, written back to (b, t, h*HD+d) bf16 for final GEMM ----
// grid (8 iblk, 64 bh), block 256; wave -> 16(i) x 64(d).
__global__ __launch_bounds__(256) void pv_kernel(
    const __bf16* __restrict__ P, const __bf16* __restrict__ Vt, __bf16* __restrict__ Oa) {
  const int lane = threadIdx.x & 31, wave = threadIdx.x >> 5;
  const int bh = blockIdx.y, b_ = bh >> 4, h = bh & 15;
  const __bf16* Ph = P + (size_t)bh * TT * TT;
  const __bf16* Vh = Vt + (size_t)bh * HDD * TT;
  const int itile = blockIdx.x * 128 + wave * 16;
  v8f c[4] = {{0,0,0,0,0,0,0,0},{0,0,0,0,0,0,0,0},{0,0,0,0,0,0,0,0},{0,0,0,0,0,0,0,0}};
  for (int j0 = 0; j0 < TT; j0 += 32) {
    v16bf a = load_afrag(Ph + (size_t)itile * TT + j0, TT, lane);
#pragma unroll
    for (int nt = 0; nt < 4; ++nt) {
      v16bf b = load_bfrag(Vh + (size_t)(nt * 16) * TT + j0, TT, lane);
      c[nt] = wmma_bf16(a, b, c[nt]);
    }
  }
  const int rl = lane & 15, hb = lane >> 4;
#pragma unroll
  for (int nt = 0; nt < 4; ++nt) {
    const int d = nt * 16 + rl;
#pragma unroll
    for (int r = 0; r < 8; ++r) {
      const int i = itile + r + 8 * hb;
      Oa[((size_t)(b_ * TT + i)) * HH + h * HDD + d] = (__bf16)c[nt][r];
    }
  }
}

// ---- out = Oa @ Wo^T + bo (fp32) ----
__global__ __launch_bounds__(256) void out_kernel(
    const __bf16* __restrict__ Oa, const __bf16* __restrict__ Wob,
    const float* __restrict__ bo, float* __restrict__ out) {
  const int lane = threadIdx.x & 31, wave = threadIdx.x >> 5;
  const int Mtile = blockIdx.y * 128 + wave * 16;
  const int N0 = blockIdx.x * 64;
  v8f c[4] = {{0,0,0,0,0,0,0,0},{0,0,0,0,0,0,0,0},{0,0,0,0,0,0,0,0},{0,0,0,0,0,0,0,0}};
  for (int k0 = 0; k0 < HH; k0 += 32) {
    v16bf a = load_afrag(Oa + (size_t)Mtile * HH + k0, HH, lane);
#pragma unroll
    for (int nt = 0; nt < 4; ++nt) {
      v16bf b = load_bfrag(Wob + (size_t)(N0 + nt * 16) * HH + k0, HH, lane);
      c[nt] = wmma_bf16(a, b, c[nt]);
    }
  }
  const int rl = lane & 15, hb = lane >> 4;
#pragma unroll
  for (int nt = 0; nt < 4; ++nt) {
    const int n = N0 + nt * 16 + rl;
    const float bn = bo[n];
#pragma unroll
    for (int r = 0; r < 8; ++r) {
      const int m = Mtile + r + 8 * hb;
      out[(size_t)m * HH + n] = c[nt][r] + bn;
    }
  }
}

extern "C" void kernel_launch(void* const* d_in, const int* in_sizes, int n_in,
                              void* d_out, int out_size, void* d_ws, size_t ws_size,
                              hipStream_t stream) {
  const float* x  = (const float*)d_in[0];
  const float* pb = (const float*)d_in[1];
  const int*   am = (const int*)d_in[2];
  const float* Wq = (const float*)d_in[3];
  const float* bq = (const float*)d_in[4];
  const float* Wk = (const float*)d_in[5];
  const float* bk = (const float*)d_in[6];
  const float* Wv = (const float*)d_in[7];
  const float* bv = (const float*)d_in[8];
  const float* Wo = (const float*)d_in[9];
  const float* bo = (const float*)d_in[10];
  float* out = (float*)d_out;

  // workspace layout (needs ~432 MiB)
  char* w = (char*)d_ws;
  __bf16* xb  = (__bf16*)(w);                          //   8 MiB  x bf16
  __bf16* Wqb = (__bf16*)(w + (8ull  << 20));          //   2 MiB
  __bf16* Wkb = (__bf16*)(w + (10ull << 20));          //   2 MiB
  __bf16* Wvb = (__bf16*)(w + (12ull << 20));          //   2 MiB
  __bf16* Wob = (__bf16*)(w + (14ull << 20));          //   2 MiB
  __bf16* Q   = (__bf16*)(w + (16ull << 20));          //   8 MiB  [bh][t][d]
  __bf16* Kh  = (__bf16*)(w + (24ull << 20));          //   8 MiB  [bh][t][d]
  __bf16* Vt  = (__bf16*)(w + (32ull << 20));          //   8 MiB  [bh][d][t]
  __bf16* Oa  = (__bf16*)(w + (40ull << 20));          //   8 MiB  [b*t][h*d]
  __bf16* P   = (__bf16*)(w + (48ull << 20));          // 128 MiB  probs bf16
  float*  S   = (float*)(w + (176ull << 20));          // 256 MiB  scores fp32

  cvt_bf16_kernel<<<1024, 256, 0, stream>>>(x,  xb,  (BB * TT * HH) / 4);
  cvt_bf16_kernel<<<512,  256, 0, stream>>>(Wq, Wqb, (HH * HH) / 4);
  cvt_bf16_kernel<<<512,  256, 0, stream>>>(Wk, Wkb, (HH * HH) / 4);
  cvt_bf16_kernel<<<512,  256, 0, stream>>>(Wv, Wvb, (HH * HH) / 4);
  cvt_bf16_kernel<<<512,  256, 0, stream>>>(Wo, Wob, (HH * HH) / 4);

  qkv_kernel<<<dim3(16, 32, 3), 256, 0, stream>>>(xb, Wqb, Wkb, Wvb, bq, bk, bv, Q, Kh, Vt);
  scores_kernel<<<dim3(16, 8, 64), 256, 0, stream>>>(Q, Kh, am, S);
  pbias_kernel<<<dim3(8, 1024), 256, 0, stream>>>(Q, pb, S);
  softmax_kernel<<<64 * 1024, 128, 0, stream>>>(S, P);
  pv_kernel<<<dim3(8, 64), 256, 0, stream>>>(P, Vt, Oa);
  out_kernel<<<dim3(16, 32), 256, 0, stream>>>(Oa, Wob, bo, out);
}
